// TreePositionEmbedding_36507222016596
// MI455X (gfx1250) — matrix operand order repaired and checked by reading, not compiled, not run on hardware
//
#include <hip/hip_runtime.h>
#include <hip/hip_bf16.h>

typedef __attribute__((ext_vector_type(16))) _Float16 v16h;
typedef __attribute__((ext_vector_type(8)))  _Float16 v8h;
typedef __attribute__((ext_vector_type(2)))  _Float16 v2h;
typedef __attribute__((ext_vector_type(8)))  float    v8f;

#define B_    4
#define H_    16
#define D_    64
#define BLK_  1024
#define KV_   4096
#define CTX_  3072
#define ST_   64
#define HID_  1024
#define KC_   128      // KV chunk per iteration

__device__ __forceinline__ v16h cat8(v8h a, v8h b) {
    return __builtin_shufflevector(a, b, 0,1,2,3,4,5,6,7,8,9,10,11,12,13,14,15);
}

// 8 waves / block; each wave owns a 16-row Q tile; block owns 128 q rows of one (b,h).
__launch_bounds__(256, 1)
__global__ void attn_wmma_kernel(const float* __restrict__ q,
                                 const float* __restrict__ k,
                                 const float* __restrict__ v,
                                 const int*   __restrict__ bvid,
                                 const float* __restrict__ relbias,
                                 const int*   __restrict__ relmat,
                                 float*       __restrict__ out)
{
    __shared__ __align__(16) _Float16 k_lds[KC_ * 64];       // [key][d]     16 KB
    __shared__ __align__(16) _Float16 vt_lds[64 * KC_];      // [d][key]     16 KB
    __shared__ __align__(16) _Float16 p_lds[8 * 16 * KC_];   // per-wave P   32 KB
    __shared__ float                  lut[128 * 64];         // bias LUT     32 KB

    const float LOG2E = 1.44269504088896340736f;

    const int bx  = blockIdx.x;
    const int b   = bx >> 7;            // / (H * BLK/128)
    const int h   = (bx >> 3) & (H_ - 1);
    const int q0  = (bx & 7) * 128;

    const int tid  = threadIdx.x;
    const int wave = tid >> 5;
    const int lane = tid & 31;
    const int nl   = lane & 15;
    const int hi   = lane >> 4;
    const int kb8  = hi * 8;

    // ---- relation-bias LUT (log2e folded): lut[q_local][vk] ----
    for (int idx = tid; idx < 128 * 64; idx += 256) {
        const int row = idx >> 6;
        const int vk  = idx & 63;
        const int vq  = bvid[b * BLK_ + q0 + row] & (ST_ - 1);
        const int rel = relmat[vq * ST_ + vk];
        lut[idx] = relbias[rel * H_ + h] * LOG2E;
    }

    // ---- Q A-fragments (ISA 7.12.2 A layout); scale = log2e / sqrt(D) folded in ----
    const float qs = 0.125f * LOG2E;
    const float* qrow = q + (((size_t)(b * H_ + h)) * BLK_ + q0 + wave * 16 + nl) * D_;
    v16h aq[2];
    #pragma unroll
    for (int s = 0; s < 2; ++s) {
        #pragma unroll
        for (int j = 0; j < 8; ++j) {
            aq[s][j]     = (_Float16)(qrow[s * 32 + kb8 + j]      * qs);
            aq[s][8 + j] = (_Float16)(qrow[s * 32 + kb8 + 16 + j] * qs);
        }
    }

    // ---- ones-column B-fragment: accumulates softmax denominator via WMMA ----
    v16h onesf;
    {
        const _Float16 oz = (nl == 0) ? (_Float16)1.0f : (_Float16)0.0f;
        #pragma unroll
        for (int j = 0; j < 16; ++j) onesf[j] = oz;
    }

    // ---- online-softmax state: o[0..3] = O tiles, o[4] col0 = row sums ----
    float mrow[8];
    v8f   o[5];
    #pragma unroll
    for (int r = 0; r < 8; ++r) mrow[r] = -INFINITY;
    #pragma unroll
    for (int t = 0; t < 5; ++t)
        #pragma unroll
        for (int r = 0; r < 8; ++r) o[t][r] = 0.0f;

    const float* kbh = k + ((size_t)(b * H_ + h)) * KV_ * D_;
    const float* vbh = v + ((size_t)(b * H_ + h)) * KV_ * D_;

    for (int kc = 0; kc < KV_ / KC_; ++kc) {
        __syncthreads();  // previous chunk's LDS fragments fully consumed

        // ---- cooperative staging: K row-major f16; V transposed, pair-packed b32 ----
        {
            const int kp = (tid & 63) * 2;        // even key, thread owns keys kp,kp+1
            const int d0 = (tid >> 6) * 16;       // 16 d-values
            const float* k0 = kbh + (size_t)(kc * KC_ + kp) * D_ + d0;
            const float* v0 = vbh + (size_t)(kc * KC_ + kp) * D_ + d0;
            _Float16 ka[16], kb2[16], va[16], vb2[16];
            #pragma unroll
            for (int j = 0; j < 16; ++j) {
                ka[j]  = (_Float16)k0[j];  kb2[j] = (_Float16)k0[j + D_];
                va[j]  = (_Float16)v0[j];  vb2[j] = (_Float16)v0[j + D_];
            }
            #pragma unroll
            for (int j = 0; j < 16; ++j) {
                k_lds[(kp    ) * 64 + d0 + j] = ka[j];
                k_lds[(kp + 1) * 64 + d0 + j] = kb2[j];
            }
            #pragma unroll
            for (int j = 0; j < 16; ++j) {
                v2h pr2 = { va[j], vb2[j] };
                *(v2h*)&vt_lds[(d0 + j) * KC_ + kp] = pr2;
            }
            if (kc + 1 < KV_ / KC_) {   // global_prefetch next chunk
                __builtin_prefetch(k0 + (size_t)KC_ * D_, 0, 1);
                __builtin_prefetch(v0 + (size_t)KC_ * D_, 0, 1);
            }
        }
        __syncthreads();

        // ---- S = Q K^T : 8 N-tiles x 2 k-steps ----
        v8f sacc[8];
        #pragma unroll
        for (int t = 0; t < 8; ++t) {
            #pragma unroll
            for (int r = 0; r < 8; ++r) sacc[t][r] = 0.0f;
            #pragma unroll
            for (int s = 0; s < 2; ++s) {
                const _Float16* kp2 = &k_lds[(t * 16 + nl) * 64 + s * 32 + hi * 8];
                v16h bk = cat8(*(const v8h*)(kp2), *(const v8h*)(kp2 + 16));
                sacc[t] = __builtin_amdgcn_wmma_f32_16x16x32_f16(
                    false, aq[s], false, bk, (short)0, sacc[t], false, false);
            }
        }

        // ---- relation bias on columns [CTX, CTX+BLK) ----
        if (kc >= CTX_ / KC_) {
            const int kb0 = kc * KC_ - CTX_;
            #pragma unroll
            for (int t = 0; t < 8; ++t) {
                const int vk = bvid[b * BLK_ + kb0 + t * 16 + nl] & (ST_ - 1);
                #pragma unroll
                for (int r = 0; r < 8; ++r)
                    sacc[t][r] += lut[(wave * 16 + r + 8 * hi) * 64 + vk];
            }
        }

        // ---- running max (rows live across 16 lanes of a half-wave) ----
        float mnew[8], al[8];
        #pragma unroll
        for (int r = 0; r < 8; ++r) {
            float mx = sacc[0][r];
            #pragma unroll
            for (int t = 1; t < 8; ++t) mx = fmaxf(mx, sacc[t][r]);
            #pragma unroll
            for (int off = 1; off < 16; off <<= 1)
                mx = fmaxf(mx, __shfl_xor(mx, off, 32));
            mnew[r] = fmaxf(mrow[r], mx);
            al[r]   = __builtin_amdgcn_exp2f(mrow[r] - mnew[r]);
            mrow[r] = mnew[r];
        }

        // ---- P = 2^(S - m), staged to wave-private LDS tile ----
        _Float16* pw = &p_lds[wave * 16 * KC_];
        #pragma unroll
        for (int t = 0; t < 8; ++t)
            #pragma unroll
            for (int r = 0; r < 8; ++r) {
                const float p = __builtin_amdgcn_exp2f(sacc[t][r] - mnew[r]);
                pw[(r + 8 * hi) * KC_ + t * 16 + nl] = (_Float16)p;
            }

        // ---- rescale accumulators (incl. denominator column) ----
        #pragma unroll
        for (int t = 0; t < 5; ++t)
            #pragma unroll
            for (int r = 0; r < 8; ++r) o[t][r] *= al[r];

        // wave-private P round-trip through LDS: fence writes before A-layout reads
        asm volatile("s_wait_dscnt 0" ::: "memory");

        // ---- O += P [V | 1] : 4 k-steps x (4 D-tiles + denominator tile) ----
        #pragma unroll
        for (int s = 0; s < 4; ++s) {
            const _Float16* pr = pw + nl * KC_ + s * 32 + kb8;
            v16h ap = cat8(*(const v8h*)(pr), *(const v8h*)(pr + 16));
            #pragma unroll
            for (int dt = 0; dt < 4; ++dt) {
                const _Float16* vp = &vt_lds[(dt * 16 + nl) * KC_ + s * 32 + hi * 8];
                v16h bv = cat8(*(const v8h*)(vp), *(const v8h*)(vp + 16));
                o[dt] = __builtin_amdgcn_wmma_f32_16x16x32_f16(
                    false, ap, false, bv, (short)0, o[dt], false, false);
            }
            o[4] = __builtin_amdgcn_wmma_f32_16x16x32_f16(
                false, ap, false, onesf, (short)0, o[4], false, false);
        }
    }

    // ---- recover row sums from o[4] column 0, normalize, store ----
    float rcp[8];
    #pragma unroll
    for (int r = 0; r < 8; ++r)
        rcp[r] = 1.0f / __shfl(o[4][r], hi * 16, 32);

    float* orow = out + (((size_t)(b * H_ + h)) * BLK_ + q0 + wave * 16) * D_;
    #pragma unroll
    for (int t = 0; t < 4; ++t)
        #pragma unroll
        for (int r = 0; r < 8; ++r) {
            const int m = r + 8 * hi;
            orow[m * D_ + t * 16 + nl] = o[t][r] * rcp[r];
        }
}

// init_bias = vertex_embedding[vid] : one block per (b, q_row), float4 copies
__global__ void init_bias_kernel(const int* __restrict__ bvid,
                                 const float* __restrict__ vemb,
                                 float* __restrict__ out2)
{
    const int row = blockIdx.x;                 // b*BLK + q_row
    const int vid = bvid[row] & (ST_ - 1);
    const float4* src = (const float4*)(vemb + (size_t)vid * HID_);
    float4*       dst = (float4*)(out2 + (size_t)row * HID_);
    for (int i = threadIdx.x; i < HID_ / 4; i += blockDim.x) dst[i] = src[i];
}

extern "C" void kernel_launch(void* const* d_in, const int* in_sizes, int n_in,
                              void* d_out, int out_size, void* d_ws, size_t ws_size,
                              hipStream_t stream) {
    const float* q       = (const float*)d_in[0];
    const float* k       = (const float*)d_in[1];
    const float* v       = (const float*)d_in[2];
    const int*   bvid    = (const int*)d_in[3];
    const float* vemb    = (const float*)d_in[4];
    const float* relbias = (const float*)d_in[5];
    const int*   relmat  = (const int*)d_in[6];

    float* out  = (float*)d_out;
    float* out2 = out + (size_t)B_ * H_ * BLK_ * D_;

    attn_wmma_kernel<<<dim3(B_ * H_ * (BLK_ / 128)), dim3(256), 0, stream>>>(
        q, k, v, bvid, relbias, relmat, out);
    init_bias_kernel<<<dim3(B_ * BLK_), dim3(256), 0, stream>>>(bvid, vemb, out2);
}
